// NormLayer_81063212744812
// MI455X (gfx1250) — matrix operand order repaired and checked by reading, not compiled, not run on hardware
//
#include <hip/hip_runtime.h>
#include <math.h>

#define NROWS 16384
#define NDIM  128
#define BK    64
#define KSTR  136   // lK row stride in halfs (128 + 8 pad -> 272B, bank-rotating)
#define VSTR  72    // lVT row stride in halfs (64 + 8 pad -> 144B, bank-rotating)

typedef _Float16 v8h  __attribute__((ext_vector_type(8)));
typedef _Float16 v16h __attribute__((ext_vector_type(16)));
typedef float    v8f  __attribute__((ext_vector_type(8)));

// 16 consecutive halfs (K = sel..sel+15 contiguous; unused here)
__device__ __forceinline__ v16h ld16_contig(const _Float16* p) {
  v8h a = *(const v8h*)p;
  v8h b = *(const v8h*)(p + 8);
  v16h r;
#pragma unroll
  for (int i = 0; i < 8; ++i) { r[i] = a[i]; r[i + 8] = b[i]; }
  return r;
}

// A-operand per-lane layout: halves 0..7 = K[sel..sel+7], halves 8..15 = K[sel+16..sel+23]
__device__ __forceinline__ v16h ld16_split(const _Float16* p) {
  v8h a = *(const v8h*)p;
  v8h b = *(const v8h*)(p + 16);
  v16h r;
#pragma unroll
  for (int i = 0; i < 8; ++i) { r[i] = a[i]; r[i + 8] = b[i]; }
  return r;
}

// ---------- pass 1: row L2 norms (reciprocal, torch-F.normalize clamp) ----------
__global__ void nl_prep_norm(const float* __restrict__ x, float* __restrict__ rnrm) {
  int row = blockIdx.x;
  int d = threadIdx.x;               // 128 threads = 4 waves
  float v = x[row * NDIM + d];
  float ss = v * v;
#pragma unroll
  for (int m = 16; m >= 1; m >>= 1) ss += __shfl_xor(ss, m, 32);
  __shared__ float part[4];
  if ((threadIdx.x & 31) == 0) part[threadIdx.x >> 5] = ss;
  __syncthreads();
  if (threadIdx.x == 0) {
    float tot = part[0] + part[1] + part[2] + part[3];
    rnrm[row] = 1.0f / fmaxf(sqrtf(tot), 1e-12f);
  }
}

// ---------- pass 2: pack Q = norm_x (f16, row-major) and VT = x^T (f16) ----------
__global__ void nl_prep_pack(const float* __restrict__ x, const float* __restrict__ rnrm,
                             _Float16* __restrict__ Qh, _Float16* __restrict__ VTh) {
  __shared__ _Float16 tile[32][33];
  int rt = blockIdx.x * 32;          // row tile base
  int ct = blockIdx.y * 32;          // dim tile base
  int tx = threadIdx.x, ty = threadIdx.y;   // 32 x 8
#pragma unroll
  for (int i = 0; i < 4; ++i) {
    int r = rt + ty + i * 8;
    int c = ct + tx;
    float v = x[(size_t)r * NDIM + c];
    Qh[(size_t)r * NDIM + c] = (_Float16)(v * rnrm[r]);
    tile[ty + i * 8][tx] = (_Float16)v;
  }
  __syncthreads();
#pragma unroll
  for (int i = 0; i < 4; ++i) {
    int d = ct + ty + i * 8;
    int row = rt + tx;
    VTh[(size_t)d * NROWS + row] = tile[tx][ty + i * 8];
  }
}

// ---------- pass 3: fused flash-attention (fixed-shift softmax) + residual + LayerNorm ----------
__global__ void __launch_bounds__(128)
nl_attn_ln(const float* __restrict__ x, const _Float16* __restrict__ Qh,
           const _Float16* __restrict__ VTh, const float* __restrict__ gamma,
           const float* __restrict__ beta, float* __restrict__ out) {
  __shared__ _Float16 lK[BK * KSTR];      // key block, row-major [64][128]+pad
  __shared__ _Float16 lVT[NDIM * VSTR];   // V^T block, [128 dims][64 keys]+pad

  const int tid  = threadIdx.x;
  const int wid  = tid >> 5;
  const int lane = tid & 31;
  const int lr   = lane & 15;
  const int sela = (lane & 16) ? 8 : 0;    // A-operand K-offset
  const int selb = (lane & 16) ? 16 : 0;   // unused; B operands use sela
  const int qbase = blockIdx.x * 64 + wid * 16;

  // Q rows resident in registers (B-operand, split K layout matching A), lane holds row qbase+lr
  v16h qreg[4];
#pragma unroll
  for (int kb = 0; kb < 4; ++kb)
    qreg[kb] = ld16_split(Qh + (size_t)(qbase + lr) * NDIM + kb * 32 + sela);

  v8f acc[8];
#pragma unroll
  for (int dt = 0; dt < 8; ++dt) acc[dt] = (v8f){};
  float lsum = 0.0f;                        // softmax denominator (per query lr)

  for (int jb = 0; jb < NROWS; jb += BK) {
    // cooperative LDS fill: 128 threads x 16 x 16B
#pragma unroll
    for (int i = 0; i < 8; ++i) {
      int c = tid + i * 128;               // 0..1023 chunks of 8 halfs
      int row = c >> 4;                    // 0..63
      int col = (c & 15) * 8;              // 0..120
      *(v8h*)(lK + row * KSTR + col) = *(const v8h*)(Qh + (size_t)(jb + row) * NDIM + col);
    }
#pragma unroll
    for (int i = 0; i < 8; ++i) {
      int c = tid + i * 128;
      int row = c >> 3;                    // 0..127 (dim)
      int col = (c & 7) * 8;               // 0..56  (key offset)
      *(v8h*)(lVT + row * VSTR + col) = *(const v8h*)(VTh + (size_t)row * NROWS + jb + col);
    }
    __syncthreads();

#pragma unroll
    for (int g = 0; g < 2; ++g) {          // two 32-key groups
      // S^T tiles: D = Kblk x Q^T  (A = key rows, B = query rows)
      v8f st0 = (v8f){}, st1 = (v8f){};
#pragma unroll
      for (int kb = 0; kb < 4; ++kb) {
        v16h a0 = ld16_split(lK + (g * 32 + 0  + lr) * KSTR + kb * 32 + sela);
        st0 = __builtin_amdgcn_wmma_f32_16x16x32_f16(false, a0, false, qreg[kb],
                                                     (short)0, st0, false, false);
        v16h a1 = ld16_split(lK + (g * 32 + 16 + lr) * KSTR + kb * 32 + sela);
        st1 = __builtin_amdgcn_wmma_f32_16x16x32_f16(false, a1, false, qreg[kb],
                                                     (short)0, st1, false, false);
      }
      // softmax numerators: cosine sim <= 1 ==> fixed shift exp(s-1), no rescaling.
      // C-layout of S^T converts lane-locally into the A-operand layout for P*V.
      v16h p;
      float ls = 0.0f;
#pragma unroll
      for (int r = 0; r < 8; ++r) {
        float e0 = __expf(st0[r] - 1.0f);
        float e1 = __expf(st1[r] - 1.0f);
        p[r]     = (_Float16)e0;
        p[r + 8] = (_Float16)e1;
        ls += e0 + e1;
      }
      lsum += ls + __shfl_xor(ls, 16, 32); // lane pair (q, q+16) holds same query
      // O += P * V  (B from V^T: split key layout matching P)
#pragma unroll
      for (int dt = 0; dt < 8; ++dt) {
        v16h bv = ld16_split(lVT + (dt * 16 + lr) * VSTR + g * 32 + sela);
        acc[dt] = __builtin_amdgcn_wmma_f32_16x16x32_f16(false, p, false, bv,
                                                         (short)0, acc[dt], false, false);
      }
    }
    __syncthreads();
  }

  // broadcast denominators into O accumulator layout (query = r + 8*hi)
  float linv[8];
#pragma unroll
  for (int r = 0; r < 8; ++r)
    linv[r] = 1.0f / __shfl(lsum, r + sela, 32);

  // y = 2x - x_neg, overwrite accumulators
#pragma unroll
  for (int dt = 0; dt < 8; ++dt) {
    int d = dt * 16 + lr;
#pragma unroll
    for (int r = 0; r < 8; ++r) {
      int q = qbase + r + sela;
      float xn = acc[dt][r] * linv[r];
      acc[dt][r] = 2.0f * x[(size_t)q * NDIM + d] - xn;
    }
  }

  // LayerNorm over D=128: in-lane over 8 dim-tiles + 16-lane tree (stays within half)
#pragma unroll
  for (int r = 0; r < 8; ++r) {
    float s1 = 0.0f, s2 = 0.0f;
#pragma unroll
    for (int dt = 0; dt < 8; ++dt) { float v = acc[dt][r]; s1 += v; s2 += v * v; }
#pragma unroll
    for (int m = 8; m >= 1; m >>= 1) { s1 += __shfl_xor(s1, m, 32); s2 += __shfl_xor(s2, m, 32); }
    float mu  = s1 * (1.0f / 128.0f);
    float var = s2 * (1.0f / 128.0f) - mu * mu;
    float rs  = rsqrtf(var + 1e-5f);
    int q = qbase + r + sela;
#pragma unroll
    for (int dt = 0; dt < 8; ++dt) {
      int d = dt * 16 + lr;
      out[(size_t)q * NDIM + d] = (acc[dt][r] - mu) * rs * gamma[d] + beta[d];
    }
  }
}

extern "C" void kernel_launch(void* const* d_in, const int* in_sizes, int n_in,
                              void* d_out, int out_size, void* d_ws, size_t ws_size,
                              hipStream_t stream) {
  (void)in_sizes; (void)n_in; (void)out_size; (void)ws_size;
  const float* x     = (const float*)d_in[0];
  const float* gamma = (const float*)d_in[1];
  const float* beta  = (const float*)d_in[2];
  float* out = (float*)d_out;

  _Float16* Qh  = (_Float16*)d_ws;                          // 4 MB: norm_x f16
  _Float16* VTh = Qh + (size_t)NROWS * NDIM;                // 4 MB: x^T f16
  float*    rnrm = (float*)(VTh + (size_t)NROWS * NDIM);    // 64 KB

  nl_prep_norm<<<NROWS, 128, 0, stream>>>(x, rnrm);
  nl_prep_pack<<<dim3(NROWS / 32, NDIM / 32), dim3(32, 8), 0, stream>>>(x, rnrm, Qh, VTh);
  nl_attn_ln<<<NROWS / 64, 128, 0, stream>>>(x, Qh, VTh, gamma, beta, out);
}
